// LinearizedContext_73778948211486
// MI455X (gfx1250) — matrix operand order repaired and checked by reading, not compiled, not run on hardware
//
#include <hip/hip_runtime.h>
#include <hip/hip_bf16.h>

typedef __attribute__((ext_vector_type(16))) _Float16 v16h;
typedef __attribute__((ext_vector_type(8)))  float    v8f;

#define NROI   1024
#define NCLS   151
#define EMBED  200
#define POSD   128
#define FUSEIN 328
#define KPAD1  352          // 328 padded to 32
#define KT1    (KPAD1/32)   // 11
#define FUSEOUT 4096
#define KT2    (FUSEOUT/32) // 128
#define NPAD2  160          // 151 padded to 16
#define BN_EPS 1e-5f
#define NMS_T  0.3f

// ---------------- workspace layout (bytes, all 256B aligned) ----------------
static constexpr size_t WS_FUSED16 = 0;                               // [1024*352] f16
static constexpr size_t WS_PACKA1  = WS_FUSED16 + (size_t)NROI*KPAD1*2;
static constexpr size_t WS_PACKB1  = WS_PACKA1  + (size_t)NROI*KPAD1*2;
static constexpr size_t WS_PREREP16= WS_PACKB1  + (size_t)KPAD1*FUSEOUT*2;
static constexpr size_t WS_PACKA2  = WS_PREREP16+ (size_t)NROI*FUSEOUT*2;
static constexpr size_t WS_PACKB2  = WS_PACKA2  + (size_t)NROI*FUSEOUT*2;
static constexpr size_t WS_PROBS   = WS_PACKB2  + (size_t)FUSEOUT*NPAD2*2;
static constexpr size_t WS_MASKED  = WS_PROBS   + (size_t)NROI*NCLS*4;

// d_out layout: dists [1024*151] | preds [1024] | pre_rep [1024*4096]
static constexpr size_t OUT_DISTS  = 0;
static constexpr size_t OUT_PREDS  = (size_t)NROI*NCLS;
static constexpr size_t OUT_PREREP = OUT_PREDS + NROI;

// ============ stage 1: softmax-embed + pos-embed + fusion BN -> f16 =========
__global__ void fuse_embed_pos_kernel(
    const float* __restrict__ logits, const float* __restrict__ box_priors,
    const float* __restrict__ embed_w,
    const float* __restrict__ pbn_g, const float* __restrict__ pbn_b,
    const float* __restrict__ pbn_m, const float* __restrict__ pbn_v,
    const float* __restrict__ pos_w, const float* __restrict__ pos_bias,
    const float* __restrict__ fbn_g, const float* __restrict__ fbn_b,
    const float* __restrict__ fbn_m, const float* __restrict__ fbn_v,
    _Float16* __restrict__ fused16)
{
    __shared__ float sm[NCLS];
    __shared__ float red[256];
    __shared__ float fused[FUSEIN];
    const int r = blockIdx.x, tid = threadIdx.x;

    float v = (tid < NCLS) ? logits[r*NCLS + tid] : -3.4e38f;
    red[tid] = v; __syncthreads();
    for (int s = 128; s > 0; s >>= 1) { if (tid < s) red[tid] = fmaxf(red[tid], red[tid+s]); __syncthreads(); }
    const float mx = red[0]; __syncthreads();
    float e = (tid < NCLS) ? __expf(v - mx) : 0.f;
    red[tid] = e; __syncthreads();
    for (int s = 128; s > 0; s >>= 1) { if (tid < s) red[tid] += red[tid+s]; __syncthreads(); }
    const float inv = 1.f / red[0];
    if (tid < NCLS) sm[tid] = e * inv;
    __syncthreads();

    if (tid < EMBED) {                       // obj_embed = softmax @ embed_w
        float acc = 0.f;
        for (int c = 0; c < NCLS; ++c) acc += sm[c] * embed_w[c*EMBED + tid];
        fused[tid] = acc;
    }
    if (tid >= EMBED && tid < FUSEIN) {      // pos = relu(BN4(cs) @ W + b)
        const int p = tid - EMBED;
        const float x1 = box_priors[r*4+0], y1 = box_priors[r*4+1];
        const float x2 = box_priors[r*4+2], y2 = box_priors[r*4+3];
        const float w = x2 - x1 + 1.f, h = y2 - y1 + 1.f;
        const float cs[4] = { x1 + 0.5f*w, y1 + 0.5f*h, w, h };
        float acc = pos_bias[p];
        for (int q = 0; q < 4; ++q) {
            const float t = pbn_g[q] * (cs[q] - pbn_m[q]) * rsqrtf(pbn_v[q] + BN_EPS) + pbn_b[q];
            acc += t * pos_w[q*POSD + p];
        }
        fused[tid] = fmaxf(acc, 0.f);
    }
    __syncthreads();

    for (int f = tid; f < KPAD1; f += 256) { // fusion BN, f16 cast, zero pad
        float y = 0.f;
        if (f < FUSEIN)
            y = fbn_g[f] * (fused[f] - fbn_m[f]) * rsqrtf(fbn_v[f] + BN_EPS) + fbn_b[f];
        fused16[(size_t)r*KPAD1 + f] = (_Float16)y;
    }
}

// =========== fragment packers (ISA 16-bit A 16x32 / B 32x16 layouts) ========
// A packed: [mtile][ktile][lane 0..31][16 halfs]; lane<16 holds K {0..7,16..23}
__global__ void packA_kernel(const _Float16* __restrict__ src, _Float16* __restrict__ dst,
                             int Mtiles, int KT)
{
    const size_t total = (size_t)Mtiles * KT * 32 * 16;
    const int Kpad = KT * 32;
    for (size_t t = (size_t)blockIdx.x*blockDim.x + threadIdx.x; t < total;
         t += (size_t)gridDim.x*blockDim.x) {
        const int h    = (int)(t & 15);
        const int lane = (int)((t >> 4) & 31);
        const int kt   = (int)((t >> 9) % KT);
        const int mt   = (int)(t / ((size_t)KT * 512));
        const int m    = mt*16 + (lane & 15);
        const int koff = (lane >> 4) * 8;
        const int kk   = (h < 8) ? (h + koff) : (h + 8 + koff);
        dst[t] = src[(size_t)m*Kpad + kt*32 + kk];
    }
}

// B packed: [ntile][ktile][lane 0..31][16 halfs]; lane<16 holds K {0..7,16..23}
__global__ void packB_kernel(const float* __restrict__ src, _Float16* __restrict__ dst,
                             int Ksrc, int Nsrc, int Ntiles, int KT)
{
    const size_t total = (size_t)Ntiles * KT * 32 * 16;
    for (size_t t = (size_t)blockIdx.x*blockDim.x + threadIdx.x; t < total;
         t += (size_t)gridDim.x*blockDim.x) {
        const int h    = (int)(t & 15);
        const int lane = (int)((t >> 4) & 31);
        const int kt   = (int)((t >> 9) % KT);
        const int nt   = (int)(t / ((size_t)KT * 512));
        const int n    = nt*16 + (lane & 15);
        const int k    = kt*32 + ((h < 8) ? (h + (lane >> 4) * 8) : (h + 8 + (lane >> 4) * 8));
        float v = 0.f;
        if (k < Ksrc && n < Nsrc) v = src[(size_t)k*Nsrc + n];
        dst[t] = (_Float16)v;
    }
}

// ==================== WMMA GEMM: one 16x16 C tile per wave ==================
__global__ void wmma_gemm_kernel(const _Float16* __restrict__ Apack,
                                 const _Float16* __restrict__ Bpack,
                                 const float* __restrict__ bias,
                                 float* __restrict__ out,
                                 int Mtiles, int Ntiles, int KT,
                                 int ldOut, int nCols, int relu)
{
    const int wave = (blockIdx.x * blockDim.x + threadIdx.x) >> 5;
    const int lane = threadIdx.x & 31;
    if (wave >= Mtiles * Ntiles) return;          // wave-uniform exit
    const int mt = wave % Mtiles;
    const int nt = wave / Mtiles;

    const _Float16* Ab = Apack + ((size_t)mt*KT)*512 + lane*16;
    const _Float16* Bb = Bpack + ((size_t)nt*KT)*512 + lane*16;

    v8f c = {};
    for (int kt = 0; kt < KT; ++kt) {
        const v16h a = *reinterpret_cast<const v16h*>(Ab + (size_t)kt*512);
        const v16h b = *reinterpret_cast<const v16h*>(Bb + (size_t)kt*512);
        if (kt + 1 < KT) {                        // global_prefetch_b8 next tiles
            __builtin_prefetch(Ab + (size_t)(kt+1)*512, 0, 1);
            __builtin_prefetch(Bb + (size_t)(kt+1)*512, 0, 1);
        }
        c = __builtin_amdgcn_wmma_f32_16x16x32_f16(
                false, a, false, b, (short)0, c, false, false);
    }

    // C layout: VGPR r -> row r + 8*(lane>=16), col = lane&15
    const int col = nt*16 + (lane & 15);
    const int rhi = (lane >> 4) * 8;
    if (col < nCols) {
        const float bv = bias[col];
        #pragma unroll
        for (int r = 0; r < 8; ++r) {
            const int row = mt*16 + r + rhi;
            float v = c[r] + bv;
            if (relu) v = fmaxf(v, 0.f);
            out[(size_t)row*ldOut + col] = v;
        }
    }
}

// ================== f32 -> f16 row-major (pre_rep for GEMM2 A) ==============
__global__ void f32_to_f16_kernel(const float* __restrict__ src, _Float16* __restrict__ dst, size_t n)
{
    for (size_t t = (size_t)blockIdx.x*blockDim.x + threadIdx.x; t < n;
         t += (size_t)gridDim.x*blockDim.x)
        dst[t] = (_Float16)src[t];
}

// ======================== row softmax over 151 logits =======================
__global__ void softmax_rows_kernel(const float* __restrict__ dists, float* __restrict__ probs)
{
    __shared__ float red[256];
    const int r = blockIdx.x, tid = threadIdx.x;
    float v = (tid < NCLS) ? dists[r*NCLS + tid] : -3.4e38f;
    red[tid] = v; __syncthreads();
    for (int s = 128; s > 0; s >>= 1) { if (tid < s) red[tid] = fmaxf(red[tid], red[tid+s]); __syncthreads(); }
    const float mx = red[0]; __syncthreads();
    float e = (tid < NCLS) ? __expf(v - mx) : 0.f;
    red[tid] = e; __syncthreads();
    for (int s = 128; s > 0; s >>= 1) { if (tid < s) red[tid] += red[tid+s]; __syncthreads(); }
    if (tid < NCLS) probs[r*NCLS + tid] = e / red[0];
}

// ============ per-class NMS: bitonic sort + greedy suppression ==============
__global__ void nms_kernel(const float* __restrict__ probs,
                           const float* __restrict__ boxes_per_cls,
                           float* __restrict__ masked)
{
    const int c = blockIdx.x + 1;               // classes 1..150
    const int tid = threadIdx.x;                // 256 threads
    __shared__ float sc[NROI];
    __shared__ int   si[NROI];
    __shared__ float bx[NROI][4];
    __shared__ unsigned char sup[NROI];

    for (int t = tid; t < NROI; t += 256) { sc[t] = probs[(size_t)t*NCLS + c]; si[t] = t; }
    __syncthreads();

    // bitonic sort, descending by (score, then original index ascending)
    for (int k = 2; k <= NROI; k <<= 1) {
        for (int j = k >> 1; j > 0; j >>= 1) {
            for (int s = 0; s < NROI/256; ++s) {
                const int t = tid + s*256;
                const int p = t ^ j;
                if (p > t) {
                    const bool descBlk = ((t & k) == 0);
                    const float s0 = sc[t], s1 = sc[p];
                    const int   i0 = si[t], i1 = si[p];
                    const bool before = (s0 > s1) || (s0 == s1 && i0 < i1);
                    if (descBlk ? !before : before) {
                        sc[t] = s1; sc[p] = s0; si[t] = i1; si[p] = i0;
                    }
                }
            }
            __syncthreads();
        }
    }

    for (int t = tid; t < NROI; t += 256) {
        const float* bp = boxes_per_cls + ((size_t)si[t]*NCLS + c)*4;
        bx[t][0] = bp[0]; bx[t][1] = bp[1]; bx[t][2] = bp[2]; bx[t][3] = bp[3];
        sup[t] = 0;
    }
    __syncthreads();

    for (int i = 0; i < NROI - 1; ++i) {
        if (!sup[i]) {
            const float ax1 = bx[i][0], ay1 = bx[i][1], ax2 = bx[i][2], ay2 = bx[i][3];
            const float aarea = (ax2 - ax1 + 1.f) * (ay2 - ay1 + 1.f);
            for (int t = tid; t < NROI; t += 256) {
                if (t > i && !sup[t]) {
                    const float ix1 = fmaxf(ax1, bx[t][0]), iy1 = fmaxf(ay1, bx[t][1]);
                    const float ix2 = fminf(ax2, bx[t][2]), iy2 = fminf(ay2, bx[t][3]);
                    const float iw = fmaxf(ix2 - ix1 + 1.f, 0.f);
                    const float ih = fmaxf(iy2 - iy1 + 1.f, 0.f);
                    const float inter = iw * ih;
                    const float barea = (bx[t][2]-bx[t][0]+1.f) * (bx[t][3]-bx[t][1]+1.f);
                    if (inter / (aarea + barea - inter) > NMS_T) sup[t] = 1;
                }
            }
        }
        __syncthreads();
    }

    for (int t = tid; t < NROI; t += 256) {
        const size_t o = (size_t)si[t]*NCLS + c;
        masked[o] = sup[t] ? 0.f : probs[o];
    }
}

// ========================= argmax over fg classes ===========================
__global__ void preds_kernel(const float* __restrict__ masked, float* __restrict__ preds)
{
    const int n = blockIdx.x*blockDim.x + threadIdx.x;
    if (n >= NROI) return;
    float best = masked[(size_t)n*NCLS + 1];
    int bi = 0;
    for (int c = 2; c < NCLS; ++c) {
        const float v = masked[(size_t)n*NCLS + c];
        if (v > best) { best = v; bi = c - 1; }
    }
    preds[n] = (float)(bi + 1);
}

// ============================== launcher ====================================
extern "C" void kernel_launch(void* const* d_in, const int* in_sizes, int n_in,
                              void* d_out, int out_size, void* d_ws, size_t ws_size,
                              hipStream_t stream) {
    const float* obj_logits  = (const float*)d_in[1];
    const float* box_priors  = (const float*)d_in[3];
    const float* boxes_pcls  = (const float*)d_in[4];
    const float* obj_embed_w = (const float*)d_in[5];
    const float* pbn_g = (const float*)d_in[6],  *pbn_b = (const float*)d_in[7];
    const float* pbn_m = (const float*)d_in[8],  *pbn_v = (const float*)d_in[9];
    const float* pos_w = (const float*)d_in[10], *pos_b = (const float*)d_in[11];
    const float* fbn_g = (const float*)d_in[12], *fbn_b = (const float*)d_in[13];
    const float* fbn_m = (const float*)d_in[14], *fbn_v = (const float*)d_in[15];
    const float* fus_w = (const float*)d_in[16], *fus_b = (const float*)d_in[17];
    const float* dec_w = (const float*)d_in[18], *dec_b = (const float*)d_in[19];

    char* ws = (char*)d_ws;
    _Float16* fused16  = (_Float16*)(ws + WS_FUSED16);
    _Float16* packA1   = (_Float16*)(ws + WS_PACKA1);
    _Float16* packB1   = (_Float16*)(ws + WS_PACKB1);
    _Float16* prerep16 = (_Float16*)(ws + WS_PREREP16);
    _Float16* packA2   = (_Float16*)(ws + WS_PACKA2);
    _Float16* packB2   = (_Float16*)(ws + WS_PACKB2);
    float*    probs    = (float*)(ws + WS_PROBS);
    float*    masked   = (float*)(ws + WS_MASKED);

    float* out      = (float*)d_out;
    float* dists    = out + OUT_DISTS;
    float* preds    = out + OUT_PREDS;
    float* pre_rep  = out + OUT_PREREP;

    // 1) fuse embeddings + both BNs -> f16 [1024, 352]
    fuse_embed_pos_kernel<<<NROI, 256, 0, stream>>>(
        obj_logits, box_priors, obj_embed_w,
        pbn_g, pbn_b, pbn_m, pbn_v, pos_w, pos_b,
        fbn_g, fbn_b, fbn_m, fbn_v, fused16);

    // 2) pack operands for GEMM1 (M=1024, K=352, N=4096)
    packA_kernel<<<256, 256, 0, stream>>>(fused16, packA1, NROI/16, KT1);
    packB_kernel<<<1024, 256, 0, stream>>>(fus_w, packB1, FUSEIN, FUSEOUT, FUSEOUT/16, KT1);

    // 3) GEMM1 + bias + ReLU -> obj_pre_rep (f32, d_out)
    {
        const int waves = (NROI/16) * (FUSEOUT/16);
        wmma_gemm_kernel<<<waves/8, 256, 0, stream>>>(
            packA1, packB1, fus_b, pre_rep,
            NROI/16, FUSEOUT/16, KT1, FUSEOUT, FUSEOUT, 1);
    }

    // 4) pre_rep -> f16, pack for GEMM2 (M=1024, K=4096, N=151->160)
    f32_to_f16_kernel<<<2048, 256, 0, stream>>>(pre_rep, prerep16, (size_t)NROI*FUSEOUT);
    packA_kernel<<<2048, 256, 0, stream>>>(prerep16, packA2, NROI/16, KT2);
    packB_kernel<<<512, 256, 0, stream>>>(dec_w, packB2, FUSEOUT, NCLS, NPAD2/16, KT2);

    // 5) GEMM2 + bias -> obj_dists2 (f32, d_out)
    {
        const int waves = (NROI/16) * (NPAD2/16);
        wmma_gemm_kernel<<<waves/8, 256, 0, stream>>>(
            packA2, packB2, dec_b, dists,
            NROI/16, NPAD2/16, KT2, NCLS, NCLS, 0);
    }

    // 6) softmax -> probs
    softmax_rows_kernel<<<NROI, 256, 0, stream>>>(dists, probs);

    // 7) per-class NMS (classes 1..150), write masked probs
    nms_kernel<<<NCLS - 1, 256, 0, stream>>>(probs, boxes_pcls, masked);

    // 8) argmax over fg classes -> obj_preds
    preds_kernel<<<(NROI + 255)/256, 256, 0, stream>>>(masked, preds);
}
